// DistributedB_25907242729685
// MI455X (gfx1250) — matrix operand order repaired and checked by reading, not compiled, not run on hardware
//
#include <hip/hip_runtime.h>
#include <hip/hip_bf16.h>
#include <math.h>

typedef __attribute__((ext_vector_type(16))) __bf16 v16bf;
typedef __attribute__((ext_vector_type(8)))  float  v8f;
typedef __attribute__((ext_vector_type(4)))  unsigned int uint4v;
typedef __attribute__((ext_vector_type(4)))  unsigned int u32x4;
typedef __attribute__((ext_vector_type(8)))  int i32x8;
typedef __attribute__((ext_vector_type(4)))  int i32x4;

#define DIMX 1024
#define KSUB 4
#define SUBD 256
#define NTOK 16384
#define EPSV 1e-6f

__device__ __forceinline__ unsigned short f2bf(float f) {
  unsigned int u = __float_as_uint(f);
  u += 0x7FFFu + ((u >> 16) & 1u);   // round-to-nearest-even
  return (unsigned short)(u >> 16);
}

// ---------------- TDM: issue a 2D bf16 tile load (tileD0 x tileD1) into LDS ----------------
// LDS layout produced: row-major, 64B of data per row + 16B pad -> matches [rows][40] ushort.
// D# per CDNA5 ISA.
__device__ __forceinline__ void tdm_load_tile(const unsigned short* gbase, unsigned ldsOff,
                                              unsigned tensorD0 /*K elems*/, unsigned tensorD1 /*rows*/,
                                              unsigned tileD0, unsigned tileD1) {
  unsigned long long ga = (unsigned long long)(uintptr_t)gbase;
  u32x4 g0;
  g0[0] = 1u;                                            // count=1, user mode, no gather
  g0[1] = ldsOff;                                        // lds_addr (bytes)
  g0[2] = (unsigned)(ga & 0xFFFFFFFFu);                  // global_addr[31:0]
  g0[3] = (unsigned)((ga >> 32) & 0x1FFFFFFu) | (2u << 30); // global_addr[56:32] | type=2
  i32x8 g1;
  // data_size=1 (2B) | pad_enable | pad_interval=3 (16 DW) | pad_amount=3 (4 DW)
  g1[0] = (int)((1u << 16) | (1u << 20) | (3u << 22) | (3u << 25));
  g1[1] = (int)((tensorD0 & 0xFFFFu) << 16);             // tensor_dim0[15:0] @bits48
  g1[2] = (int)(((tensorD0 >> 16) & 0xFFFFu) | ((tensorD1 & 0xFFFFu) << 16));
  g1[3] = (int)(((tensorD1 >> 16) & 0xFFFFu) | ((tileD0 & 0xFFFFu) << 16)); // tile_dim0 @bits112
  g1[4] = (int)(tileD1 & 0xFFFFu);                       // tile_dim1 ; tile_dim2=0
  g1[5] = (int)tensorD0;                                 // tensor_dim0_stride[31:0] (= K)
  g1[6] = 0;                                             // stride hi / dim1_stride lo
  g1[7] = 0;
  i32x4 z4 = {0, 0, 0, 0};
#if __clang_major__ >= 23
  i32x8 z8 = {0, 0, 0, 0, 0, 0, 0, 0};
  __builtin_amdgcn_tensor_load_to_lds(g0, g1, z4, z4, z8, 0);
#else
  __builtin_amdgcn_tensor_load_to_lds(g0, g1, z4, z4, 0);
#endif
}

// ---------------- pack / weight-prep kernels ----------------

__global__ void pack_xh_kernel(const float* __restrict__ x, const float* __restrict__ h,
                               unsigned short* __restrict__ xh) {
  size_t idx = (size_t)blockIdx.x * blockDim.x + threadIdx.x; // NTOK*2048
  size_t n = idx >> 11; int c = (int)(idx & 2047);
  float v = (c < DIMX) ? x[n * DIMX + c] : h[n * DIMX + (c - DIMX)];
  xh[idx] = f2bf(v);
}

__global__ void w1a_pack_kernel(const float* __restrict__ W1, unsigned short* __restrict__ Wc1) {
  size_t idx = (size_t)blockIdx.x * blockDim.x + threadIdx.x; // 1024*1024
  size_t r = idx >> 10; size_t c = idx & 1023;
  Wc1[r * 2048 + c] = f2bf(W1[r * 2048 + c]);
}

// Weff[k,j,d] = sum_e W1[k,j,1024+e] * Wv[k,e,d]  -> Wcomb1 row k*256+j, col 1024+d
__global__ void weff_kernel(const float* __restrict__ W1, const float* __restrict__ Wv,
                            unsigned short* __restrict__ Wc1) {
  int k = blockIdx.z;
  int d0 = blockIdx.x * 16, j0 = blockIdx.y * 16;
  int tx = threadIdx.x, ty = threadIdx.y;
  __shared__ float As[16][17], Bs[16][17];
  const float* A = W1 + (size_t)k * SUBD * 2048 + DIMX;
  const float* B = Wv + (size_t)k * DIMX * DIMX;
  float acc = 0.f;
  for (int e0 = 0; e0 < DIMX; e0 += 16) {
    As[ty][tx] = A[(size_t)(j0 + ty) * 2048 + e0 + tx];
    Bs[ty][tx] = B[(size_t)(e0 + ty) * DIMX + d0 + tx];
    __syncthreads();
#pragma unroll
    for (int e = 0; e < 16; ++e) acc += As[ty][e] * Bs[e][tx];
    __syncthreads();
  }
  Wc1[(size_t)(k * SUBD + j0 + ty) * 2048 + DIMX + d0 + tx] = f2bf(acc);
}

__global__ void w2cat_kernel(const float* __restrict__ W2, unsigned short* __restrict__ Wc2) {
  size_t idx = (size_t)blockIdx.x * blockDim.x + threadIdx.x; // 4*1024*256
  int k = (int)(idx >> 18); int d = (int)((idx >> 8) & 1023); int j = (int)(idx & 255);
  Wc2[(size_t)d * 1024 + k * 256 + j] = f2bf(W2[((size_t)k * 1024 + d) * 256 + j]);
}

// G[k,d,j] = sum_e Wl[d, k*1024+e] * W2[k,e,j] -> Wcomb2 row 1024+d, col k*256+j
__global__ void gcat_kernel(const float* __restrict__ Wl, const float* __restrict__ W2,
                            unsigned short* __restrict__ Wc2) {
  int k = blockIdx.z;
  int j0 = blockIdx.x * 16, d0 = blockIdx.y * 16;
  int tx = threadIdx.x, ty = threadIdx.y; // tx=j, ty=d
  __shared__ float As[16][17], Bs[16][17];
  float acc = 0.f;
  for (int e0 = 0; e0 < DIMX; e0 += 16) {
    As[ty][tx] = Wl[(size_t)(d0 + ty) * (KSUB * DIMX) + k * DIMX + e0 + tx];
    Bs[ty][tx] = W2[((size_t)k * DIMX + e0 + ty) * SUBD + j0 + tx];
    __syncthreads();
#pragma unroll
    for (int e = 0; e < 16; ++e) acc += As[ty][e] * Bs[e][tx];
    __syncthreads();
  }
  Wc2[(size_t)(DIMX + d0 + ty) * 1024 + k * SUBD + j0 + tx] = f2bf(acc);
}

__global__ void wlsum_kernel(const float* __restrict__ Wl, unsigned short* __restrict__ Ws) {
  size_t idx = (size_t)blockIdx.x * blockDim.x + threadIdx.x; // 1024*1024
  int d = (int)(idx >> 10), e = (int)(idx & 1023);
  float s = 0.f;
#pragma unroll
  for (int k = 0; k < KSUB; ++k) s += Wl[(size_t)d * 4096 + k * 1024 + e];
  Ws[idx] = f2bf(s);
}

// ---------------- WMMA bf16 GEMM with TDM double-buffered tile pipeline ----------------
// C[M,N'] = A[M,K] @ B[N',K]^T.  MODE 0: f32 -> Cf.  MODE 1: silu -> bf16 -> Cb.
template<int MODE>
__global__ __launch_bounds__(256) void gemm_bf16_kernel(
    const unsigned short* __restrict__ A, const unsigned short* __restrict__ B,
    float* __restrict__ Cf, unsigned short* __restrict__ Cb, int Kdim, int ldc) {
  // [2 buffers][128 rows][32 data + 8 pad shorts]; TDM pad fields reproduce this layout.
  __shared__ unsigned short lds_a[2][128][40];
  __shared__ unsigned short lds_b[2][128][40];

  const int tid  = threadIdx.x;
  const int lane = tid & 31;
  const int wave = tid >> 5;
  const int half = lane >> 4;
  const int l16  = lane & 15;
  const int wm   = wave & 3;  // 4 waves along M -> 32 rows each
  const int wn   = wave >> 2; // 2 waves along N -> 64 cols each
  const int blockM = blockIdx.y * 128;
  const int blockN = blockIdx.x * 128;
  const unsigned rowsA = gridDim.y * 128u;
  const unsigned rowsB = gridDim.x * 128u;

  v8f acc[2][4];
  const v8f vzero = {0.f, 0.f, 0.f, 0.f, 0.f, 0.f, 0.f, 0.f};
#pragma unroll
  for (int i = 0; i < 2; ++i)
#pragma unroll
    for (int j = 0; j < 4; ++j) acc[i][j] = vzero;

  const unsigned short* aTile = A + (size_t)blockM * Kdim;
  const unsigned short* bTile = B + (size_t)blockN * Kdim;
  const unsigned ldsA0 = (unsigned)(uintptr_t)&lds_a[0][0][0];
  const unsigned ldsA1 = (unsigned)(uintptr_t)&lds_a[1][0][0];
  const unsigned ldsB0 = (unsigned)(uintptr_t)&lds_b[0][0][0];
  const unsigned ldsB1 = (unsigned)(uintptr_t)&lds_b[1][0][0];

  const int nIter = Kdim >> 5;

  // prologue: wave 0 kicks off TDM for tile pair 0 (TDM ignores EXEC; branch keeps it per-wave)
  if (tid < 32) {
    tdm_load_tile(aTile, ldsA0, (unsigned)Kdim, rowsA, 32u, 128u);
    tdm_load_tile(bTile, ldsB0, (unsigned)Kdim, rowsB, 32u, 128u);
  }

  for (int it = 0; it < nIter; ++it) {
    const int buf = it & 1;
    if (tid < 32) __builtin_amdgcn_s_wait_tensorcnt(0); // tile pair `it` landed in LDS
    __syncthreads();                                    // all waves: data ready, other buf free
    if (tid < 32 && (it + 1) < nIter) {                 // async-issue next pair into other buf
      const int nb = (it + 1) & 1;
      const unsigned short* aN = aTile + (size_t)(it + 1) * 32;
      const unsigned short* bN = bTile + (size_t)(it + 1) * 32;
      tdm_load_tile(aN, nb ? ldsA1 : ldsA0, (unsigned)Kdim, rowsA, 32u, 128u);
      tdm_load_tile(bN, nb ? ldsB1 : ldsB0, (unsigned)Kdim, rowsB, 32u, 128u);
      if (it + 2 < nIter) {                             // warm L2 for the pair after that
        __builtin_prefetch(aTile + (size_t)(it + 2) * 32, 0, 1);
        __builtin_prefetch(bTile + (size_t)(it + 2) * 32, 0, 1);
      }
    }

    // A fragment (16x32 bf16): elems 0..7 -> K = 8*half+0..7 ; elems 8..15 -> K = 16+8*half+0..7
    v16bf afrag[2], bfrag[4];
#pragma unroll
    for (int mt = 0; mt < 2; ++mt) {
      const unsigned short* p = &lds_a[buf][wm * 32 + mt * 16 + l16][half * 8];
      union { uint4v u[2]; v16bf v; } f;
      f.u[0] = *reinterpret_cast<const uint4v*>(p);
      f.u[1] = *reinterpret_cast<const uint4v*>(p + 16);
      afrag[mt] = f.v;
    }
    // B fragment (32x16 bf16): elems 0..7 -> K = 8*half+0..7 ; elems 8..15 -> K = 16+8*half+0..7, N = lane%16
#pragma unroll
    for (int nt = 0; nt < 4; ++nt) {
      const unsigned short* p = &lds_b[buf][wn * 64 + nt * 16 + l16][half * 8];
      union { uint4v u[2]; v16bf v; } f;
      f.u[0] = *reinterpret_cast<const uint4v*>(p);
      f.u[1] = *reinterpret_cast<const uint4v*>(p + 16);
      bfrag[nt] = f.v;
    }
#pragma unroll
    for (int mt = 0; mt < 2; ++mt)
#pragma unroll
      for (int nt = 0; nt < 4; ++nt)
        acc[mt][nt] = __builtin_amdgcn_wmma_f32_16x16x32_bf16(
            false, afrag[mt], false, bfrag[nt], (short)0, acc[mt][nt], false, false);
  }

  // C/D layout: VGPR j -> m = j + 8*half ; n = lane%16
  const int mBase = blockM + wm * 32;
  const int nBase = blockN + wn * 64;
#pragma unroll
  for (int mt = 0; mt < 2; ++mt) {
#pragma unroll
    for (int nt = 0; nt < 4; ++nt) {
      int col = nBase + nt * 16 + l16;
#pragma unroll
      for (int j = 0; j < 8; ++j) {
        int row = mBase + mt * 16 + half * 8 + j;
        float v = acc[mt][nt][j];
        if (MODE == 0) {
          Cf[(size_t)row * ldc + col] = v;
        } else {
          float s = v / (1.f + __expf(-v));   // silu
          Cb[(size_t)row * ldc + col] = f2bf(s);
        }
      }
    }
  }
}

// ---------------- rmsnorm / final ----------------

__global__ __launch_bounds__(256) void rmsnorm1_kernel(const float* __restrict__ C2,
                                                       const float* __restrict__ g,
                                                       unsigned short* __restrict__ u1b) {
  int n = blockIdx.x;
  const float* y1 = C2 + (size_t)n * 2048 + 1024;
  __shared__ float red[256];
  float vals[4]; float ss = 0.f;
#pragma unroll
  for (int i = 0; i < 4; ++i) {
    int d = threadIdx.x + i * 256;
    vals[i] = y1[d];
    ss += vals[i] * vals[i];
  }
  red[threadIdx.x] = ss; __syncthreads();
  for (int s = 128; s > 0; s >>= 1) {
    if (threadIdx.x < s) red[threadIdx.x] += red[threadIdx.x + s];
    __syncthreads();
  }
  float rinv = rsqrtf(red[0] / 1024.f + EPSV);
#pragma unroll
  for (int i = 0; i < 4; ++i) {
    int d = threadIdx.x + i * 256;
    u1b[(size_t)n * 1024 + d] = f2bf(vals[i] * rinv * g[d]);
  }
}

__global__ __launch_bounds__(256) void final_kernel(const float* __restrict__ C2,
                                                    const float* __restrict__ z,
                                                    const float* __restrict__ g,
                                                    float* __restrict__ out) {
  int n = blockIdx.x;
  const float* psum = C2 + (size_t)n * 2048;
  const float* y1   = psum + 1024;
  const float* zr   = z + (size_t)n * 1024;
  __shared__ float red[256];
  float y1v[4], y2v[4], gv[4], ps[4];
  float ss1 = 0.f, ss2 = 0.f;
#pragma unroll
  for (int i = 0; i < 4; ++i) {
    int d = threadIdx.x + i * 256;
    y1v[i] = y1[d]; ps[i] = psum[d]; gv[i] = g[d];
    y2v[i] = y1v[i] + zr[d];
    ss1 += y1v[i] * y1v[i];
    ss2 += y2v[i] * y2v[i];
  }
  red[threadIdx.x] = ss1; __syncthreads();
  for (int s = 128; s > 0; s >>= 1) {
    if (threadIdx.x < s) red[threadIdx.x] += red[threadIdx.x + s];
    __syncthreads();
  }
  float rinv1 = rsqrtf(red[0] / 1024.f + EPSV);
  __syncthreads();
  red[threadIdx.x] = ss2; __syncthreads();
  for (int s = 128; s > 0; s >>= 1) {
    if (threadIdx.x < s) red[threadIdx.x] += red[threadIdx.x + s];
    __syncthreads();
  }
  float rinv2 = rsqrtf(red[0] / 1024.f + EPSV);
#pragma unroll
  for (int i = 0; i < 4; ++i) {
    int d = threadIdx.x + i * 256;
    float u1 = y1v[i] * rinv1 * gv[i];
    float u2 = y2v[i] * rinv2 * gv[i];
    out[(size_t)n * 1024 + d] = tanhf(ps[i] * 0.25f + u1 + u2);
  }
}

// ---------------- launch ----------------

extern "C" void kernel_launch(void* const* d_in, const int* in_sizes, int n_in,
                              void* d_out, int out_size, void* d_ws, size_t ws_size,
                              hipStream_t stream) {
  (void)in_sizes; (void)n_in; (void)out_size; (void)ws_size;
  const float* x  = (const float*)d_in[0];
  const float* h  = (const float*)d_in[1];
  const float* Wv = (const float*)d_in[2];
  const float* W1 = (const float*)d_in[3];
  const float* W2 = (const float*)d_in[4];
  const float* Wl = (const float*)d_in[5];
  const float* g  = (const float*)d_in[6];
  float* out = (float*)d_out;

  char* ws = (char*)d_ws;
  unsigned short* xh  = (unsigned short*)(ws);                      // 64 MB  [N,2048] bf16
  unsigned short* mid = (unsigned short*)(ws + ((size_t)64 << 20)); // 32 MB  [N,1024] bf16
  float* C2           = (float*)(ws + ((size_t)96 << 20));          // 128 MB [N,2048] f32 (psum|y1)
  unsigned short* u1b = (unsigned short*)(ws + ((size_t)224 << 20));// 32 MB  [N,1024] bf16
  float* z            = (float*)(ws + ((size_t)256 << 20));         // 64 MB  [N,1024] f32
  unsigned short* Wc1 = (unsigned short*)(ws + ((size_t)320 << 20));// 4 MB   [1024,2048] bf16
  unsigned short* Wc2 = (unsigned short*)(ws + ((size_t)324 << 20));// 4 MB   [2048,1024] bf16
  unsigned short* Wls = (unsigned short*)(ws + ((size_t)328 << 20));// 2 MB   [1024,1024] bf16

  // activation + weight prep (small)
  pack_xh_kernel<<<(NTOK * 2048) / 256, 256, 0, stream>>>(x, h, xh);
  w1a_pack_kernel<<<(1024 * 1024) / 256, 256, 0, stream>>>(W1, Wc1);
  weff_kernel<<<dim3(64, 16, KSUB), dim3(16, 16), 0, stream>>>(W1, Wv, Wc1);
  w2cat_kernel<<<(KSUB * 1024 * 256) / 256, 256, 0, stream>>>(W2, Wc2);
  gcat_kernel<<<dim3(16, 64, KSUB), dim3(16, 16), 0, stream>>>(Wl, W2, Wc2);
  wlsum_kernel<<<(1024 * 1024) / 256, 256, 0, stream>>>(Wl, Wls);

  // GEMM1: mid = silu(xh @ Wc1^T)            [N,1024], K=2048
  gemm_bf16_kernel<1><<<dim3(1024 / 128, NTOK / 128), 256, 0, stream>>>(xh, Wc1, nullptr, mid, 2048, 1024);
  // GEMM2: C2 = mid @ Wc2^T = [psum | y1]    [N,2048], K=1024
  gemm_bf16_kernel<0><<<dim3(2048 / 128, NTOK / 128), 256, 0, stream>>>(mid, Wc2, C2, nullptr, 1024, 2048);
  // u1 = rmsnorm(y1)
  rmsnorm1_kernel<<<NTOK, 256, 0, stream>>>(C2, g, u1b);
  // GEMM3: z = u1 @ Wlsum^T                  [N,1024], K=1024
  gemm_bf16_kernel<0><<<dim3(1024 / 128, NTOK / 128), 256, 0, stream>>>(u1b, Wls, z, nullptr, 1024, 1024);
  // out = tanh(psum/4 + u1 + rmsnorm(y1+z))
  final_kernel<<<NTOK, 256, 0, stream>>>(C2, z, g, out);
}
